// QINCoInferenceStepEncoder_62775241998556
// MI455X (gfx1250) — matrix operand order repaired and checked by reading, not compiled, not run on hardware
//
#include <hip/hip_runtime.h>
#include <hip/hip_bf16.h>
#include <math.h>

// ---------------- problem constants ----------------
#define B_BASE 512
#define F_IN   16
#define F_OUT  16
#define DIM    128
#define DH     256
#define KCB    4096
#define AA     32
#define MM     2
#define LL     2
#define BTOT   (B_BASE * F_IN)        // 8192
#define CAND   (BTOT * AA)            // 262144

typedef __bf16 bf16_t;
typedef bf16_t v16bf __attribute__((ext_vector_type(16)));
typedef float  v8f   __attribute__((ext_vector_type(8)));

union FragBF { v16bf v; uint4 q[2]; };

// ---------------- WMMA fragment helpers ----------------
// A operand (16x32 bf16): lane holds row m = lane&15; one contiguous 16-elem
// chunk: K = half*16 + [0..15], half = lane>>4.
__device__ __forceinline__ v16bf load_a_frag(const bf16_t* A, int stride, int m0,
                                             int k0, int lane) {
  int m    = m0 + (lane & 15);
  int half = (lane >> 4) & 1;
  const bf16_t* p = A + (size_t)m * stride + k0 + half * 16;
  FragBF f;
  f.q[0] = *(const uint4*)p;         // elems 0..7
  f.q[1] = *(const uint4*)(p + 8);   // elems 8..15
  return f.v;
}

// B operand from pre-packed weights: [ktile][ntile][lane][16 bf16] contiguous.
__device__ __forceinline__ v16bf load_b_frag_packed(const bf16_t* Wp, int NT,
                                                    int kt, int nt, int lane) {
  const bf16_t* p = Wp + (((size_t)(kt * NT + nt) * 32 + lane) << 4);
  FragBF f;
  f.q[0] = *(const uint4*)p;
  f.q[1] = *(const uint4*)(p + 8);
  return f.v;
}

// B operand straight from a row-major [N][K] bf16 matrix where each "column"
// of B is a contiguous row (codebook case): lane holds col n = lane&15,
// K = (lane>>4)*16 + e contiguous.
__device__ __forceinline__ v16bf load_b_frag_rows(const bf16_t* Wrows, int rowStride,
                                                  int n0, int k0, int lane) {
  int n    = n0 + (lane & 15);
  int half = (lane >> 4) & 1;
  const bf16_t* p = Wrows + (size_t)n * rowStride + k0 + half * 16;
  FragBF f;
  f.q[0] = *(const uint4*)p;
  f.q[1] = *(const uint4*)(p + 8);
  return f.v;
}

__device__ __forceinline__ v8f wmma_bf16(v16bf a, v16bf b, v8f c) {
  return __builtin_amdgcn_wmma_f32_16x16x32_bf16(false, a, false, b,
                                                 (short)0, c, false, false);
}

// C/D layout: VGPR v -> row m = v + 8*half, col n = n0 + (lane&15)
__device__ __forceinline__ void store_acc_bf16(bf16_t* dst, int stride, int m0,
                                               int n0, int lane, v8f acc, bool relu) {
  int n = n0 + (lane & 15);
  int half = (lane >> 4) & 1;
#pragma unroll
  for (int v = 0; v < 8; ++v) {
    float val = acc[v];
    if (relu) val = fmaxf(val, 0.0f);
    dst[(size_t)(m0 + v + half * 8) * stride + n] = (bf16_t)val;
  }
}

// ---------------- prep kernels ----------------
__global__ __launch_bounds__(256) void prep_xt_kernel(const float* __restrict__ x,
                                                      const float* __restrict__ xhat,
                                                      bf16_t* __restrict__ xt_bf,
                                                      bf16_t* __restrict__ xh_bf) {
  int i = blockIdx.x * 256 + threadIdx.x;
  if (i >= BTOT * DIM) return;
  int d  = i & (DIM - 1);
  int b  = i >> 7;
  int bb = b >> 4;
  float xv = x[bb * DIM + d];
  float hv = xhat[i];
  xt_bf[i] = (bf16_t)(xv - hv);
  xh_bf[i] = (bf16_t)hv;
}

__global__ __launch_bounds__(256) void prep_cb_kernel(const float* __restrict__ cb,
                                                      const float* __restrict__ cbrq,
                                                      bf16_t* __restrict__ cb_bf,
                                                      bf16_t* __restrict__ cbrq_bf) {
  int i = blockIdx.x * 256 + threadIdx.x;
  if (i >= KCB * DIM) return;
  cb_bf[i]   = (bf16_t)cb[i];
  cbrq_bf[i] = (bf16_t)cbrq[i];
}

__global__ __launch_bounds__(256) void cbsq_kernel(const float* __restrict__ cbrq,
                                                   float* __restrict__ cb_sq) {
  int k = blockIdx.x * 256 + threadIdx.x;
  if (k >= KCB) return;
  float s = 0.f;
#pragma unroll 4
  for (int d = 0; d < DIM; ++d) {
    float v = cbrq[k * DIM + d];
    s += v * v;
  }
  cb_sq[k] = s;
}

// Pack row-major f32 W[K][N] into B-fragment order [kt][nt][lane][e] bf16.
__global__ __launch_bounds__(256) void pack_weight_kernel(const float* __restrict__ W,
                                                          bf16_t* __restrict__ out,
                                                          int K, int N) {
  int i = blockIdx.x * 256 + threadIdx.x;
  if (i >= K * N) return;
  int e    = i & 15;
  int lane = (i >> 4) & 31;
  int t    = i >> 9;
  int NT   = N >> 4;
  int nt   = t % NT;
  int kt   = t / NT;
  int k = kt * 32 + (lane >> 4) * 16 + e;
  int n = nt * 16 + (lane & 15);
  out[i] = (bf16_t)W[(size_t)k * N + n];
}

// ---------------- stage 1: dists = cb_sq - 2 * xt @ cbrq^T ----------------
__global__ __launch_bounds__(256) void dist_kernel(const bf16_t* __restrict__ xt,
                                                   const bf16_t* __restrict__ cbrq,
                                                   const float* __restrict__ cb_sq,
                                                   bf16_t* __restrict__ dists) {
  int lane = threadIdx.x & 31;
  int wave = threadIdx.x >> 5;   // 0..7, each wave covers 512 codes
  int m0   = blockIdx.x * 16;    // 512 blocks -> 8192 rows

  v16bf afr[4];
#pragma unroll
  for (int kt = 0; kt < 4; ++kt) afr[kt] = load_a_frag(xt, DIM, m0, kt * 32, lane);

  int nbase = wave * 512;
  for (int nt = 0; nt < 32; ++nt) {
    int n0 = nbase + nt * 16;
    v8f acc = {};
#pragma unroll
    for (int kt = 0; kt < 4; ++kt) {
      v16bf b = load_b_frag_rows(cbrq, DIM, n0, kt * 32, lane);
      acc = wmma_bf16(afr[kt], b, acc);
    }
    int n    = n0 + (lane & 15);
    int half = (lane >> 4) & 1;
    float csq = cb_sq[n];
#pragma unroll
    for (int v = 0; v < 8; ++v) {
      int m = m0 + v + half * 8;
      dists[(size_t)m * KCB + n] = (bf16_t)(csq - 2.0f * acc[v]);
    }
  }
}

// ---------------- stage 1b: top-32 shortlist per row (register-resident) ----
// Candidate i = slot*256 + tid lives in thread tid's register slot.
__global__ __launch_bounds__(256) void topk32_kernel(const bf16_t* __restrict__ dists,
                                                     int* __restrict__ top_codes) {
  __shared__ float wv[8];
  __shared__ int   wi[8];
  __shared__ int   besti;
  int row = blockIdx.x;
  int tid = threadIdx.x;
  float rv[16];
#pragma unroll
  for (int s = 0; s < 16; ++s)
    rv[s] = (float)dists[(size_t)row * KCB + s * 256 + tid];

  for (int t = 0; t < AA; ++t) {
    // local min over register slots (ties -> lowest slot -> lowest index)
    float bv = rv[0];
    int   bs = 0;
#pragma unroll
    for (int s = 1; s < 16; ++s)
      if (rv[s] < bv) { bv = rv[s]; bs = s; }
    int bi = bs * 256 + tid;
    // wave32 shuffle reduction
#pragma unroll
    for (int off = 16; off > 0; off >>= 1) {
      float ov = __shfl_down(bv, (unsigned)off);
      int   oi = __shfl_down(bi, (unsigned)off);
      if (ov < bv || (ov == bv && oi < bi)) { bv = ov; bi = oi; }
    }
    int wave = tid >> 5;
    if ((tid & 31) == 0) { wv[wave] = bv; wi[wave] = bi; }
    __syncthreads();
    if (tid == 0) {
      float fb = wv[0]; int fi = wi[0];
#pragma unroll
      for (int w = 1; w < 8; ++w)
        if (wv[w] < fb || (wv[w] == fb && wi[w] < fi)) { fb = wv[w]; fi = wi[w]; }
      top_codes[row * AA + t] = fi;
      besti = fi;
    }
    __syncthreads();
    int fi = besti;
    if ((fi & 255) == tid) rv[fi >> 8] = INFINITY;
    // next tid==0 write to besti is separated by the next round's first barrier
  }
}

// ---------------- stage 2: fused candidate MLP (all WMMA) ----------------
// 1 WG = 64 threads = 2 waves = 32 candidate rows.
// A fragments for each GEMM stage are cached in registers across n-tiles.
__global__ __launch_bounds__(64) void mlp_kernel(
    const float* __restrict__ x,
    const float* __restrict__ in_b,  const float* __restrict__ cc_b,
    const float* __restrict__ rb_b1, const float* __restrict__ rb_b2,
    const float* __restrict__ out_b,
    const bf16_t* __restrict__ cb_bf, const bf16_t* __restrict__ xh_bf,
    const int* __restrict__ top_codes,
    const bf16_t* __restrict__ w_in,  const bf16_t* __restrict__ w_cc,
    const bf16_t* __restrict__ w_r1,  const bf16_t* __restrict__ w_r2,
    const bf16_t* __restrict__ w_out,
    bf16_t* __restrict__ r_out, float* __restrict__ d2_out) {
  __shared__ bf16_t smem[24576];            // 48 KB
  bf16_t* cwin = smem;                      // [32][128] gathered codewords
  bf16_t* xh   = smem + 4096;               // [32][128] xhat per row
  bf16_t* hbuf = smem + 8192;               // [32][256]
  bf16_t* cbuf = smem + 16384;              // [32][256]
  float*  rbuf = (float*)(smem + 8192);     // alias hbuf region after last use

  int tid  = threadIdx.x;
  int lane = tid & 31;
  int wave = tid >> 5;
  int m0   = wave * 16;
  int n15  = lane & 15;
  int half = (lane >> 4) & 1;
  int c0   = blockIdx.x * 32;               // candidate row base

  // gather codewords + xhat (uint4 = 8 bf16 per copy; 512 copies per buffer)
  for (int i = tid; i < 32 * 16; i += 64) {
    int r    = i >> 4;          // row in block
    int col8 = i & 15;
    int c  = c0 + r;
    int bb = c >> 9;            // / 512
    int f  = (c >> 5) & 15;
    int a  = c & 31;
    int code = top_codes[((bb << 4) + f) * AA + a];
    ((uint4*)cwin)[i] = ((const uint4*)(cb_bf + (size_t)code * DIM))[col8];
    ((uint4*)xh)[i]   = ((const uint4*)(xh_bf + (size_t)((bb << 4) + f) * DIM))[col8];
  }
  __syncthreads();

  v16bf afr[8];
  v16bf afrX[4];

  // ---- GEMM1: hbuf = cwin @ in_w + in_b   (K=128, N=256) ----
#pragma unroll
  for (int kt = 0; kt < 4; ++kt) afr[kt] = load_a_frag(cwin, DIM, m0, kt * 32, lane);
  for (int nt = 0; nt < 16; ++nt) {
    float bias = in_b[nt * 16 + n15];
    v8f acc;
#pragma unroll
    for (int v = 0; v < 8; ++v) acc[v] = bias;
#pragma unroll
    for (int kt = 0; kt < 4; ++kt)
      acc = wmma_bf16(afr[kt], load_b_frag_packed(w_in, 16, kt, nt, lane), acc);
    store_acc_bf16(hbuf, DH, m0, nt * 16, lane, acc, false);
  }
  __syncthreads();

  // ---- GEMM2: cbuf = [hbuf, xh] @ cc_w + cc_b (K=256+128, N=256) ----
#pragma unroll
  for (int kt = 0; kt < 8; ++kt) afr[kt]  = load_a_frag(hbuf, DH, m0, kt * 32, lane);
#pragma unroll
  for (int kt = 0; kt < 4; ++kt) afrX[kt] = load_a_frag(xh, DIM, m0, kt * 32, lane);
  for (int nt = 0; nt < 16; ++nt) {
    float bias = cc_b[nt * 16 + n15];
    v8f acc;
#pragma unroll
    for (int v = 0; v < 8; ++v) acc[v] = bias;
#pragma unroll
    for (int kt = 0; kt < 8; ++kt)
      acc = wmma_bf16(afr[kt], load_b_frag_packed(w_cc, 16, kt, nt, lane), acc);
#pragma unroll
    for (int kt = 8; kt < 12; ++kt)
      acc = wmma_bf16(afrX[kt - 8], load_b_frag_packed(w_cc, 16, kt, nt, lane), acc);
    store_acc_bf16(cbuf, DH, m0, nt * 16, lane, acc, false);
  }
  __syncthreads();

  // ---- residual blocks: cbuf += relu(cbuf@w1+b1)@w2 + b2 ----
  for (int l = 0; l < LL; ++l) {
    const bf16_t* W1 = w_r1 + (size_t)l * DH * DH;
    const bf16_t* W2 = w_r2 + (size_t)l * DH * DH;
#pragma unroll
    for (int kt = 0; kt < 8; ++kt) afr[kt] = load_a_frag(cbuf, DH, m0, kt * 32, lane);
    for (int nt = 0; nt < 16; ++nt) {
      float bias = rb_b1[l * DH + nt * 16 + n15];
      v8f acc;
#pragma unroll
      for (int v = 0; v < 8; ++v) acc[v] = bias;
#pragma unroll
      for (int kt = 0; kt < 8; ++kt)
        acc = wmma_bf16(afr[kt], load_b_frag_packed(W1, 16, kt, nt, lane), acc);
      store_acc_bf16(hbuf, DH, m0, nt * 16, lane, acc, true);   // fused ReLU
    }
    __syncthreads();
#pragma unroll
    for (int kt = 0; kt < 8; ++kt) afr[kt] = load_a_frag(hbuf, DH, m0, kt * 32, lane);
    for (int nt = 0; nt < 16; ++nt) {
      int n = nt * 16 + n15;
      float bias = rb_b2[l * DH + n];
      v8f acc;
#pragma unroll
      for (int v = 0; v < 8; ++v)
        acc[v] = bias + (float)cbuf[(size_t)(m0 + v + half * 8) * DH + n];
#pragma unroll
      for (int kt = 0; kt < 8; ++kt)
        acc = wmma_bf16(afr[kt], load_b_frag_packed(W2, 16, kt, nt, lane), acc);
      store_acc_bf16(cbuf, DH, m0, nt * 16, lane, acc, false);
    }
    __syncthreads();
  }

  // ---- out: r = cbuf@out_w + out_b + cwin + xh  (K=256, N=128) ----
#pragma unroll
  for (int kt = 0; kt < 8; ++kt) afr[kt] = load_a_frag(cbuf, DH, m0, kt * 32, lane);
  for (int nt = 0; nt < 8; ++nt) {
    int n = nt * 16 + n15;
    v8f acc;
#pragma unroll
    for (int v = 0; v < 8; ++v)
      acc[v] = out_b[n] + (float)cwin[(size_t)(m0 + v + half * 8) * DIM + n];
#pragma unroll
    for (int kt = 0; kt < 8; ++kt)
      acc = wmma_bf16(afr[kt], load_b_frag_packed(w_out, 8, kt, nt, lane), acc);
#pragma unroll
    for (int v = 0; v < 8; ++v) {
      int m = m0 + v + half * 8;
      rbuf[(size_t)m * DIM + n] = acc[v] + (float)xh[(size_t)m * DIM + n];
    }
  }
  __syncthreads();

  // ---- d2 per row + write bf16 reconstructions ----
  if (tid < 32) {
    int c    = c0 + tid;
    int bb   = c >> 9;
    int kidx = c & 511;
    float rr = 0.f, xr = 0.f;
#pragma unroll 4
    for (int d = 0; d < DIM; ++d) {
      float rvl = rbuf[tid * DIM + d];
      rr += rvl * rvl;
      xr += x[bb * DIM + d] * rvl;
    }
    d2_out[bb * 512 + kidx] = rr - 2.0f * xr;
  }
  for (int i = tid; i < 32 * DIM; i += 64)
    r_out[(size_t)c0 * DIM + i] = (bf16_t)rbuf[i];
}

// ---------------- stage 3+4: per-base top-16 + regather ----------------
__global__ __launch_bounds__(256) void finalize_kernel(const float* __restrict__ d2,
                                                       const int* __restrict__ top_codes,
                                                       const int* __restrict__ codes_MBF,
                                                       const bf16_t* __restrict__ r_ws,
                                                       float* __restrict__ out) {
  __shared__ float ld[512];
  __shared__ float sv[256];
  __shared__ int   si[256];
  __shared__ int   widx[F_OUT];
  int bb  = blockIdx.x;
  int tid = threadIdx.x;
  ld[tid]       = d2[bb * 512 + tid];
  ld[tid + 256] = d2[bb * 512 + 256 + tid];
  __syncthreads();
  for (int t = 0; t < F_OUT; ++t) {
    float bv = ld[tid];        int bi = tid;
    float v2 = ld[tid + 256];
    if (v2 < bv || (v2 == bv && (tid + 256) < bi)) { bv = v2; bi = tid + 256; }
    sv[tid] = bv; si[tid] = bi;
    __syncthreads();
    for (int s = 128; s > 0; s >>= 1) {
      if (tid < s) {
        if (sv[tid + s] < sv[tid] ||
            (sv[tid + s] == sv[tid] && si[tid + s] < si[tid])) {
          sv[tid] = sv[tid + s]; si[tid] = si[tid + s];
        }
      }
      __syncthreads();
    }
    if (tid == 0) { widx[t] = si[0]; ld[si[0]] = INFINITY; }
    __syncthreads();
  }
  // xhat_next [512][16][128]
  for (int i = tid; i < F_OUT * DIM; i += 256) {
    int j = i >> 7;
    int d = i & (DIM - 1);
    int idx = widx[j];
    out[(size_t)bb * (F_OUT * DIM) + i] =
        (float)r_ws[((size_t)bb * 512 + idx) * DIM + d];
  }
  // codes_out [3][512][16] appended after xhat_next (value-cast to float)
  if (tid < F_OUT) {
    int j   = tid;
    int idx = widx[j];
    int f   = idx >> 5;
    int a   = idx & 31;
    int cg  = top_codes[((bb << 4) + f) * AA + a];
    const size_t base = (size_t)B_BASE * F_OUT * DIM;
    out[base + 0 * B_BASE * F_OUT + bb * F_OUT + j] =
        (float)codes_MBF[0 * B_BASE * F_IN + bb * F_IN + f];
    out[base + 1 * B_BASE * F_OUT + bb * F_OUT + j] =
        (float)codes_MBF[1 * B_BASE * F_IN + bb * F_IN + f];
    out[base + 2 * B_BASE * F_OUT + bb * F_OUT + j] = (float)cg;
  }
}

// ---------------- host launcher ----------------
extern "C" void kernel_launch(void* const* d_in, const int* in_sizes, int n_in,
                              void* d_out, int out_size, void* d_ws, size_t ws_size,
                              hipStream_t stream) {
  const float* x_BD      = (const float*)d_in[0];
  const float* xhat_BFD  = (const float*)d_in[1];
  const int*   codes_MBF = (const int*)  d_in[2];
  const float* codebook  = (const float*)d_in[3];
  const float* cbrq      = (const float*)d_in[4];
  const float* in_w      = (const float*)d_in[5];
  const float* in_b      = (const float*)d_in[6];
  const float* cc_w      = (const float*)d_in[7];
  const float* cc_b      = (const float*)d_in[8];
  const float* rb_w1     = (const float*)d_in[9];
  const float* rb_b1     = (const float*)d_in[10];
  const float* rb_w2     = (const float*)d_in[11];
  const float* rb_b2     = (const float*)d_in[12];
  const float* out_w     = (const float*)d_in[13];
  const float* out_b     = (const float*)d_in[14];
  float* out = (float*)d_out;

  // carve workspace (256B aligned)
  size_t off = 0;
  auto carve = [&](size_t bytes) {
    void* p = (char*)d_ws + off;
    off += (bytes + 255) & ~(size_t)255;
    return p;
  };
  bf16_t* xt_bf   = (bf16_t*)carve((size_t)BTOT * DIM * 2);
  bf16_t* xh_bf   = (bf16_t*)carve((size_t)BTOT * DIM * 2);
  bf16_t* cb_bf   = (bf16_t*)carve((size_t)KCB * DIM * 2);
  bf16_t* cbrq_bf = (bf16_t*)carve((size_t)KCB * DIM * 2);
  float*  cb_sq   = (float*) carve((size_t)KCB * 4);
  bf16_t* pw_in   = (bf16_t*)carve((size_t)DIM * DH * 2);
  bf16_t* pw_cc   = (bf16_t*)carve((size_t)(DH + DIM) * DH * 2);
  bf16_t* pw_r1   = (bf16_t*)carve((size_t)LL * DH * DH * 2);
  bf16_t* pw_r2   = (bf16_t*)carve((size_t)LL * DH * DH * 2);
  bf16_t* pw_out  = (bf16_t*)carve((size_t)DH * DIM * 2);
  int*    top_cd  = (int*)   carve((size_t)BTOT * AA * 4);
  float*  d2_ws   = (float*) carve((size_t)B_BASE * 512 * 4);
  bf16_t* dists   = (bf16_t*)carve((size_t)BTOT * KCB * 2);    // 64 MB
  bf16_t* r_ws    = (bf16_t*)carve((size_t)CAND * DIM * 2);    // 64 MB
  (void)ws_size; (void)in_sizes; (void)n_in; (void)out_size;

  // prep
  prep_xt_kernel<<<(BTOT * DIM + 255) / 256, 256, 0, stream>>>(x_BD, xhat_BFD, xt_bf, xh_bf);
  prep_cb_kernel<<<(KCB * DIM + 255) / 256, 256, 0, stream>>>(codebook, cbrq, cb_bf, cbrq_bf);
  cbsq_kernel<<<(KCB + 255) / 256, 256, 0, stream>>>(cbrq, cb_sq);

  // weight packing
  pack_weight_kernel<<<(DIM * DH + 255) / 256, 256, 0, stream>>>(in_w, pw_in, DIM, DH);
  pack_weight_kernel<<<((DH + DIM) * DH + 255) / 256, 256, 0, stream>>>(cc_w, pw_cc, DH + DIM, DH);
  for (int l = 0; l < LL; ++l) {
    pack_weight_kernel<<<(DH * DH + 255) / 256, 256, 0, stream>>>(
        rb_w1 + (size_t)l * DH * DH, pw_r1 + (size_t)l * DH * DH, DH, DH);
    pack_weight_kernel<<<(DH * DH + 255) / 256, 256, 0, stream>>>(
        rb_w2 + (size_t)l * DH * DH, pw_r2 + (size_t)l * DH * DH, DH, DH);
  }
  pack_weight_kernel<<<(DH * DIM + 255) / 256, 256, 0, stream>>>(out_w, pw_out, DH, DIM);

  // stage 1
  dist_kernel<<<BTOT / 16, 256, 0, stream>>>(xt_bf, cbrq_bf, cb_sq, dists);
  topk32_kernel<<<BTOT, 256, 0, stream>>>(dists, top_cd);

  // stage 2
  mlp_kernel<<<CAND / 32, 64, 0, stream>>>(x_BD, in_b, cc_b, rb_b1, rb_b2, out_b,
                                           cb_bf, xh_bf, top_cd,
                                           pw_in, pw_cc, pw_r1, pw_r2, pw_out,
                                           r_ws, d2_ws);

  // stage 3+4
  finalize_kernel<<<B_BASE, 256, 0, stream>>>(d2_ws, top_cd, codes_MBF, r_ws, out);
}
